// Attention_cait_10058813407918
// MI455X (gfx1250) — matrix operand order repaired and checked by reading, not compiled, not run on hardware
//
#include <hip/hip_runtime.h>
#include <hip/hip_bf16.h>

// ---------------------------------------------------------------------------
// CaiT talking-heads attention for gfx1250 (MI455X), bf16 WMMA pipeline.
//   K0: prep  - x -> bf16; w_q|w_kv, w_out -> bf16 transposed (col-major)
//   K1: qkv   - xb @ wtb -> q (pre-scaled) [b,h,n,d], k [b,h,n,d], v^T [b,h,d,n]
//   K2: attn  - fused talking-heads flash attention (2-pass recompute),
//               transposed LDS score tiles -> vector ds ops + packed f32 mixes
//   K3: proj  - attn_out @ w_out + b_out -> fp32 output
// ---------------------------------------------------------------------------

typedef __bf16 bf16_t;
typedef __attribute__((ext_vector_type(16))) __bf16 v16bf;
typedef __attribute__((ext_vector_type(8)))  __bf16 v8bf;
typedef __attribute__((ext_vector_type(4)))  __bf16 v4bf;
typedef __attribute__((ext_vector_type(8)))  float  v8f;
typedef __attribute__((ext_vector_type(4)))  float  v4f;

#define TDIM   512
#define THEADS 8
#define TDH    64
#define TB     8
#define TN     1024
#define TSCALE 0.125f   // 64^-0.5

// CDNA5 ISA 7.12.2 layouts (wave32):
//  A 16x32 bf16 : lane m = lane&15, khalf = lane>>4; element e -> K:
//                 e<8 : K = e + 8*khalf ; e>=8 : K = 16 + (e-8) + 8*khalf
//  B 32x16 bf16 : lane n = lane&15; same K order as A (this code loads A and B as K = e + 16*khalf)
//  C/D 16x16 f32: VGPR r -> (m = r + 8*khalf, n = lane&15).

__device__ __forceinline__ v8f wmma_bf16(v16bf a, v16bf b, v8f c) {
  return __builtin_amdgcn_wmma_f32_16x16x32_bf16(false, a, false, b, (short)0, c,
                                                 false, false);
}

__device__ __forceinline__ v16bf load_afrag(const bf16_t* rowk0, int khalf) {
  v8bf r0 = *(const v8bf*)(rowk0 + 16 * khalf);
  v8bf r1 = *(const v8bf*)(rowk0 + 8 + 16 * khalf);
  return __builtin_shufflevector(r0, r1, 0, 1, 2, 3, 4, 5, 6, 7,
                                 8, 9, 10, 11, 12, 13, 14, 15);
}

// ---------------------------------------------------------------------------
// K0a: x fp32 -> bf16 row-major (8192x512)
// ---------------------------------------------------------------------------
__global__ __launch_bounds__(256) void cvtx_kernel(const float* __restrict__ x,
                                                   bf16_t* __restrict__ xb) {
  const size_t i = ((size_t)blockIdx.x * 256 + threadIdx.x) * 4;
  v4f v = *(const v4f*)(x + i);
  v4bf o;
#pragma unroll
  for (int e = 0; e < 4; ++e) o[e] = (bf16_t)v[e];
  *(v4bf*)(xb + i) = o;
}

// ---------------------------------------------------------------------------
// K0b: weight transpose+convert: dst[c*512 + k] = (bf16)src[k*C + c]
// ---------------------------------------------------------------------------
__global__ __launch_bounds__(256) void wtrans_kernel(const float* __restrict__ src,
                                                     bf16_t* __restrict__ dst,
                                                     int C) {
  const int c = blockIdx.x;
  for (int k = threadIdx.x; k < TDIM; k += 256)
    dst[(size_t)c * TDIM + k] = (bf16_t)src[(size_t)k * C + c];
}

// ---------------------------------------------------------------------------
// K1: QKV projection. xb [8192,512] bf16 @ wtb^T (wtb is [1536][512] col-major).
// One wave = one 16x16 tile; grid (96 col tiles, 64 row groups) x 8 waves.
// ---------------------------------------------------------------------------
__global__ __launch_bounds__(256) void qkv_proj_kernel(
    const bf16_t* __restrict__ xb, const bf16_t* __restrict__ wtb,
    bf16_t* __restrict__ qb, bf16_t* __restrict__ kb, bf16_t* __restrict__ vbT) {
  const int wave = threadIdx.x >> 5, lane = threadIdx.x & 31;
  const int mtile = blockIdx.y * 8 + wave;  // 0..511
  const int c0 = blockIdx.x * 16;           // 0..1535 (wave-uniform)
  const int row0 = mtile * 16;
  const int m = lane & 15, khalf = lane >> 4, n = m;

  v8f acc = {};
  const bf16_t* arow = xb + (size_t)(row0 + m) * TDIM;
  const bf16_t* brow = wtb + (size_t)(c0 + n) * TDIM;
  for (int k0 = 0; k0 < TDIM; k0 += 32) {
    v16bf a = load_afrag(arow + k0, khalf);
    v16bf b = *(const v16bf*)(brow + k0 + 16 * khalf);
    acc = wmma_bf16(a, b, acc);
  }

  const int bb = row0 >> 10, nn0 = row0 & 1023;  // 16 rows never cross batch
  if (c0 < 512) {            // ---- Q (scaled), [b,h,n,d]
    const int h = c0 >> 6, d0 = c0 & 63;
    bf16_t* p = qb + ((size_t)(bb * 8 + h) * TN + nn0 + 8 * khalf) * TDH + d0 + n;
#pragma unroll
    for (int r = 0; r < 8; ++r) p[(size_t)r * TDH] = (bf16_t)(acc[r] * TSCALE);
  } else if (c0 < 1024) {    // ---- K, [b,h,n,d]
    const int cc = c0 - 512, h = cc >> 6, d0 = cc & 63;
    bf16_t* p = kb + ((size_t)(bb * 8 + h) * TN + nn0 + 8 * khalf) * TDH + d0 + n;
#pragma unroll
    for (int r = 0; r < 8; ++r) p[(size_t)r * TDH] = (bf16_t)acc[r];
  } else {                   // ---- V transposed, [b,h,d,n]; 8 consecutive n
    const int cc = c0 - 1024, h = cc >> 6, d0 = cc & 63;
    v8bf pk;
#pragma unroll
    for (int r = 0; r < 8; ++r) pk[r] = (bf16_t)acc[r];
    *(v8bf*)(vbT + ((size_t)(bb * 8 + h) * TDH + d0 + n) * TN + nn0 + 8 * khalf) = pk;
  }
}

// ---------------------------------------------------------------------------
// K2: fused talking-heads attention.  Block = (b, 16-row query tile), 8 waves,
// wave w owns head w.  Score tiles live transposed in LDS: [head][col][row],
// so C-frag stores and mix-loop loads are all ds_*_b128, and the head mixes
// run as packed f32 vector math.
// ---------------------------------------------------------------------------
__global__ __launch_bounds__(256) void attn_kernel(
    const bf16_t* __restrict__ qb, const bf16_t* __restrict__ kb,
    const bf16_t* __restrict__ vbT,
    const float* __restrict__ mix_pre, const float* __restrict__ mix_post,
    bf16_t* __restrict__ ob) {
  __shared__ float  SldsT[THEADS][32][16];  // [h][col j][row t]
  __shared__ float  PldsT[THEADS][32][16];  // [g][col j][row t]
  __shared__ bf16_t Alds[16][32][THEADS];   // [row t][col j][g'] -> see note
  __shared__ float  PRE[64], POST[64];

  const int w = threadIdx.x >> 5, lane = threadIdx.x & 31;
  if (threadIdx.x < 64) {
    PRE[threadIdx.x]  = mix_pre[threadIdx.x];
    POST[threadIdx.x] = mix_post[threadIdx.x];
  }
  const int bI = blockIdx.y;
  const int i0 = blockIdx.x * 16;
  const int m = lane & 15, khalf = lane >> 4;

  const bf16_t* Q  = qb  + ((size_t)(bI * 8 + w) * TN + i0) * TDH;
  const bf16_t* K  = kb  + (size_t)(bI * 8 + w) * TN * TDH;
  const bf16_t* VT = vbT + (size_t)(bI * 8 + w) * TDH * TN;

  v16bf qa[2];
#pragma unroll
  for (int dh = 0; dh < 2; ++dh)
    qa[dh] = load_afrag(Q + (size_t)m * TDH + dh * 32, khalf);

  float mx[16], sm[16];
#pragma unroll
  for (int t = 0; t < 16; ++t) { mx[t] = -3.0e38f; sm[t] = 0.f; }

  __syncthreads();  // PRE/POST ready

  // ---------------- pass 1: softmax statistics ----------------
  for (int j0 = 0; j0 < TN; j0 += 32) {
#pragma unroll
    for (int jt = 0; jt < 2; ++jt) {
      v8f sc = {};
#pragma unroll
      for (int dh = 0; dh < 2; ++dh) {
        v16bf kf = *(const v16bf*)(K + (size_t)(j0 + jt * 16 + m) * TDH +
                                   dh * 32 + 16 * khalf);
        sc = wmma_bf16(qa[dh], kf, sc);
      }
      // C-frag lane holds rows (8*khalf..+7) at col jt*16+m: 2x ds_store_b128
      v4f lo = {sc[0], sc[1], sc[2], sc[3]};
      v4f hi = {sc[4], sc[5], sc[6], sc[7]};
      *(v4f*)&SldsT[w][jt * 16 + m][8 * khalf]     = lo;
      *(v4f*)&SldsT[w][jt * 16 + m][8 * khalf + 4] = hi;
    }
    __syncthreads();
    // lane L owns column (j0+L): mix heads (packed f32), then online update
    v4f sacc[4] = {};
#pragma unroll
    for (int h = 0; h < THEADS; ++h) {
      const float pw = PRE[h * 8 + w];
#pragma unroll
      for (int q = 0; q < 4; ++q)
        sacc[q] += *(const v4f*)&SldsT[h][lane][q * 4] * pw;
    }
#pragma unroll
    for (int q = 0; q < 4; ++q)
#pragma unroll
      for (int e = 0; e < 4; ++e) {
        const int t = q * 4 + e;
        const float s = sacc[q][e];
        float nm = fmaxf(mx[t], s);
        sm[t] = sm[t] * __expf(mx[t] - nm) + __expf(s - nm);
        mx[t] = nm;
      }
    __syncthreads();
  }
  // merge per-lane stats -> per-row (result broadcast to all lanes)
#pragma unroll
  for (int t = 0; t < 16; ++t) {
    float rm = mx[t];
#pragma unroll
    for (int off = 16; off; off >>= 1) rm = fmaxf(rm, __shfl_xor(rm, off, 32));
    float rs = sm[t] * __expf(mx[t] - rm);
#pragma unroll
    for (int off = 16; off; off >>= 1) rs += __shfl_xor(rs, off, 32);
    mx[t] = rm;
    sm[t] = 1.0f / rs;
  }

  // ---------------- pass 2: P -> post-mix -> O += A @ V ----------------
  v8f o[4] = {};
  for (int j0 = 0; j0 < TN; j0 += 32) {
#pragma unroll
    for (int jt = 0; jt < 2; ++jt) {
      v8f sc = {};
#pragma unroll
      for (int dh = 0; dh < 2; ++dh) {
        v16bf kf = *(const v16bf*)(K + (size_t)(j0 + jt * 16 + m) * TDH +
                                   dh * 32 + 16 * khalf);
        sc = wmma_bf16(qa[dh], kf, sc);
      }
      v4f lo = {sc[0], sc[1], sc[2], sc[3]};
      v4f hi = {sc[4], sc[5], sc[6], sc[7]};
      *(v4f*)&SldsT[w][jt * 16 + m][8 * khalf]     = lo;
      *(v4f*)&SldsT[w][jt * 16 + m][8 * khalf + 4] = hi;
    }
    __syncthreads();
    {  // P = exp(mix_pre(S) - m) / l, stored transposed (vector stores)
      v4f sacc[4] = {};
#pragma unroll
      for (int h = 0; h < THEADS; ++h) {
        const float pw = PRE[h * 8 + w];
#pragma unroll
        for (int q = 0; q < 4; ++q)
          sacc[q] += *(const v4f*)&SldsT[h][lane][q * 4] * pw;
      }
#pragma unroll
      for (int q = 0; q < 4; ++q) {
        v4f pv;
#pragma unroll
        for (int e = 0; e < 4; ++e) {
          const int t = q * 4 + e;
          pv[e] = __expf(sacc[q][e] - mx[t]) * sm[t];
        }
        *(v4f*)&PldsT[w][lane][q * 4] = pv;
      }
    }
    __syncthreads();
    {  // post-softmax head mix (packed f32) -> bf16 A tile (row-major)
      v4f aacc[4] = {};
#pragma unroll
      for (int g = 0; g < THEADS; ++g) {
        const float pw = POST[g * 8 + w];
#pragma unroll
        for (int q = 0; q < 4; ++q)
          aacc[q] += *(const v4f*)&PldsT[g][lane][q * 4] * pw;
      }
      // Alds indexed [t][lane][w]: each wave owns slice w, rows t, col lane.
#pragma unroll
      for (int q = 0; q < 4; ++q)
#pragma unroll
        for (int e = 0; e < 4; ++e)
          Alds[q * 4 + e][lane][w] = (bf16_t)aacc[q][e];
    }
    __builtin_amdgcn_wave_barrier();  // Alds slice self-produced/consumed per wave
    // A-frag: element (row m, k = col): gather across cols (stride THEADS bf16)
    v16bf af;
#pragma unroll
    for (int e = 0; e < 8; ++e) af[e] = Alds[m][e + 16 * khalf][w];
#pragma unroll
    for (int e = 0; e < 8; ++e) af[8 + e] = Alds[m][8 + e + 16 * khalf][w];
#pragma unroll
    for (int ds = 0; ds < 4; ++ds) {
      v16bf vf = *(const v16bf*)(VT + (size_t)(ds * 16 + m) * TN + j0 + 16 * khalf);
      o[ds] = wmma_bf16(af, vf, o[ds]);
    }
    __syncthreads();
  }

  // store O to [b*n, inner] bf16 for the output projection
#pragma unroll
  for (int ds = 0; ds < 4; ++ds)
#pragma unroll
    for (int r = 0; r < 8; ++r)
      ob[(size_t)((bI << 10) + i0 + r + 8 * khalf) * TDIM + w * 64 + ds * 16 + m] =
          (bf16_t)o[ds][r];
}

// ---------------------------------------------------------------------------
// K3: output projection. attn_out [8192,512] bf16 @ w_out + b_out -> fp32.
// ---------------------------------------------------------------------------
__global__ __launch_bounds__(256) void out_proj_kernel(
    const bf16_t* __restrict__ ob, const bf16_t* __restrict__ wotb,
    const float* __restrict__ bout, float* __restrict__ out) {
  const int wave = threadIdx.x >> 5, lane = threadIdx.x & 31;
  const int mtile = blockIdx.y * 8 + wave;
  const int c0 = blockIdx.x * 16;
  const int row0 = mtile * 16;
  const int m = lane & 15, khalf = lane >> 4, n = m;

  v8f acc = {};
  const bf16_t* arow = ob + (size_t)(row0 + m) * TDIM;
  const bf16_t* brow = wotb + (size_t)(c0 + n) * TDIM;
  for (int k0 = 0; k0 < TDIM; k0 += 32) {
    v16bf a = load_afrag(arow + k0, khalf);
    v16bf b = *(const v16bf*)(brow + k0 + 16 * khalf);
    acc = wmma_bf16(a, b, acc);
  }
  const float bias = bout[c0 + n];
#pragma unroll
  for (int r = 0; r < 8; ++r)
    out[(size_t)(row0 + r + 8 * khalf) * TDIM + c0 + n] = acc[r] + bias;
}

// ---------------------------------------------------------------------------
extern "C" void kernel_launch(void* const* d_in, const int* in_sizes, int n_in,
                              void* d_out, int out_size, void* d_ws, size_t ws_size,
                              hipStream_t stream) {
  const float* x        = (const float*)d_in[0];
  const float* w_q      = (const float*)d_in[1];
  const float* w_kv     = (const float*)d_in[2];
  const float* mix_pre  = (const float*)d_in[3];
  const float* mix_post = (const float*)d_in[4];
  const float* w_out    = (const float*)d_in[5];
  const float* b_out    = (const float*)d_in[6];
  float* out = (float*)d_out;

  char* ws = (char*)d_ws;
  const size_t SZ = (size_t)TB * THEADS * TN * TDH * sizeof(bf16_t);  // 8 MB
  bf16_t* xb   = (bf16_t*)(ws);                 // 8 MB  [8192][512]
  bf16_t* qb   = (bf16_t*)(ws + SZ);            // 8 MB  [b,h,n,d]
  bf16_t* kb   = (bf16_t*)(ws + 2 * SZ);        // 8 MB  [b,h,n,d]
  bf16_t* vbT  = (bf16_t*)(ws + 3 * SZ);        // 8 MB  [b,h,d,n]
  bf16_t* ob   = (bf16_t*)(ws + 4 * SZ);        // 8 MB  [8192][512]
  bf16_t* wtb  = (bf16_t*)(ws + 5 * SZ);        // 1.5 MB [1536][512]
  bf16_t* wotb = wtb + (size_t)1536 * TDIM;     // 0.5 MB [512][512]

  cvtx_kernel<<<dim3(4096), 256, 0, stream>>>(x, xb);
  wtrans_kernel<<<dim3(512), 256, 0, stream>>>(w_q, wtb, 512);
  wtrans_kernel<<<dim3(1024), 256, 0, stream>>>(w_kv, wtb + (size_t)512 * TDIM, 1024);
  wtrans_kernel<<<dim3(512), 256, 0, stream>>>(w_out, wotb, 512);

  qkv_proj_kernel<<<dim3(96, 64), 256, 0, stream>>>(xb, wtb, qb, kb, vbT);
  attn_kernel<<<dim3(TN / 16, TB), 256, 0, stream>>>(qb, kb, vbT, mix_pre, mix_post, ob);
  out_proj_kernel<<<dim3(TDIM / 16, 64), 256, 0, stream>>>(ob, wotb, b_out, out);
}
